// GNN_17532056502678
// MI455X (gfx1250) — matrix operand order repaired and checked by reading, not compiled, not run on hardware
//
#include <hip/hip_runtime.h>

#define NN 100000
#define NE 1600000
#define FD 128
#define NG 64

typedef __attribute__((ext_vector_type(8)))  _Float16 v8h;
typedef __attribute__((ext_vector_type(16))) _Float16 v16h;
typedef __attribute__((ext_vector_type(8)))  float    v8f;

// ---------- small utility kernels ----------
__global__ void k_fill1(float* p, int n) {
    int i = blockIdx.x * 256 + threadIdx.x;
    if (i < n) p[i] = 1.0f;
}
__global__ void k_zero(float* p, int n) {
    int i = blockIdx.x * 256 + threadIdx.x;
    if (i < n) p[i] = 0.0f;
}
__global__ void k_deg(const int* __restrict__ dst, float* __restrict__ deg) {
    int e = blockIdx.x * 256 + threadIdx.x;
    if (e < NE) atomicAdd(&deg[dst[e]], 1.0f);
}
__global__ void k_rsqrt(float* d, int n) {
    int i = blockIdx.x * 256 + threadIdx.x;
    if (i < n) d[i] = rsqrtf(fmaxf(d[i], 1.0f));
}
__global__ void k_cast_f16(const float* __restrict__ x, _Float16* __restrict__ h, int n) {
    int i = blockIdx.x * 256 + threadIdx.x;
    if (i < n) h[i] = (_Float16)x[i];
}
// store W (FDxFD, row-major, row=k) transposed: Wt[n*FD + k] = W[k*FD + n]
__global__ void k_cast_wT(const float* __restrict__ w, _Float16* __restrict__ wt) {
    int i = blockIdx.x * 256 + threadIdx.x;   // over FD*FD
    int k = i >> 7, n = i & (FD - 1);
    wt[n * FD + k] = (_Float16)w[k * FD + n];
}
__global__ void k_copy4(const float4* __restrict__ s, float4* __restrict__ d, int n) {
    int i = blockIdx.x * 256 + threadIdx.x;
    if (i < n) d[i] = s[i];
}
__global__ void k_count(const int* __restrict__ batch, float* __restrict__ cnt) {
    int i = blockIdx.x * 256 + threadIdx.x;
    if (i < NN) atomicAdd(&cnt[batch[i]], 1.0f);
}

// ---------- WMMA GEMM: Gout[m,n] = dinv[m] * sum_k A[m,k]*Wt[n,k] ----------
// A: NN x FD f16 row-major; Wt: FD x FD f16 (transposed weights, row = output col)
// one wave per 16x16 tile, K looped 4 x 32 with v_wmma_f32_16x16x32_f16
__global__ __launch_bounds__(256)
void k_gemm_scaled(const _Float16* __restrict__ A, const _Float16* __restrict__ Wt,
                   const float* __restrict__ dinv, float* __restrict__ Gout) {
    const int wave = threadIdx.x >> 5;
    const int lane = threadIdx.x & 31;
    const int tile = blockIdx.x * 8 + wave;       // (NN/16)*(FD/16) tiles total
    const int tm = tile >> 3;                     // 0..6249
    const int tn = tile & 7;                      // 0..7
    const int m0 = tm * 16, n0 = tn * 16;
    const int hsel = lane >> 4;                   // 0 / 1 (half-wave)
    const int l15  = lane & 15;

    // A-fragment: row = m0+l15; elements 0..7 -> K = k0 + 8*hsel + j
    //             elements 8..15 -> K = k0 + 8*hsel + 16 + j
    const _Float16* arow = A  + (size_t)(m0 + l15) * FD + hsel * 8;
    // B-fragment: col = n0+l15; same K order as A (k0 + 8*hsel + j, k0 + 8*hsel + 16 + j)
    const _Float16* brow = Wt + (size_t)(n0 + l15) * FD + hsel * 8;

    v8f c = {};
#pragma unroll
    for (int k0 = 0; k0 < FD; k0 += 32) {
        v8h alo = *(const v8h*)(arow + k0);
        v8h ahi = *(const v8h*)(arow + k0 + 16);
        v16h a = __builtin_shufflevector(alo, ahi, 0,1,2,3,4,5,6,7,8,9,10,11,12,13,14,15);
        v8h blo = *(const v8h*)(brow + k0);
        v8h bhi = *(const v8h*)(brow + k0 + 16);
        v16h b = __builtin_shufflevector(blo, bhi, 0,1,2,3,4,5,6,7,8,9,10,11,12,13,14,15);
        c = __builtin_amdgcn_wmma_f32_16x16x32_f16(false, a, false, b, (short)0, c, false, false);
    }

    const int n = n0 + l15;
#pragma unroll
    for (int r = 0; r < 8; ++r) {
        const int m = m0 + r + hsel * 8;
        Gout[(size_t)m * FD + n] = dinv[m] * c[r];
    }
}

// ---------- edge scatter: agg[dst] += g[src]; one wave per edge ----------
__global__ __launch_bounds__(256)
void k_scatter(const int* __restrict__ src, const int* __restrict__ dst,
               const float* __restrict__ g, float* __restrict__ agg) {
    const int wave = threadIdx.x >> 5;
    const int lane = threadIdx.x & 31;
    const int e = blockIdx.x * 8 + wave;
    if (e >= NE) return;
    const int s = src[e], d = dst[e];
    const float4 v = *(const float4*)(g + (size_t)s * FD + lane * 4);
    float* o = agg + (size_t)d * FD + lane * 4;
    atomicAdd(o + 0, v.x);
    atomicAdd(o + 1, v.y);
    atomicAdd(o + 2, v.z);
    atomicAdd(o + 3, v.w);
}

// ---------- layer-1 epilogue: relu(dinv*agg + b1) -> f16 input of layer 2 ----------
__global__ void k_epi_relu_cast(const float* __restrict__ agg, const float* __restrict__ dinv,
                                const float* __restrict__ b, _Float16* __restrict__ hb) {
    int idx = blockIdx.x * 256 + threadIdx.x;      // over NN*FD
    int i = idx >> 7, f = idx & (FD - 1);
    float v = fmaxf(dinv[i] * agg[idx] + b[f], 0.0f);
    hb[idx] = (_Float16)v;
}

// ---------- layer-2 epilogue fused with graph pooling sums ----------
__global__ void k_epi_pool(const float* __restrict__ agg, const float* __restrict__ dinv,
                           const float* __restrict__ b, const int* __restrict__ batch,
                           float* __restrict__ pool) {
    int idx = blockIdx.x * 256 + threadIdx.x;      // over NN*FD
    int i = idx >> 7, f = idx & (FD - 1);
    float v = dinv[i] * agg[idx] + b[f];
    atomicAdd(&pool[batch[i] * FD + f], v);
}

// ---------- final: out[g] = (pool[g,:] . Wf) / cnt[g] + bf ----------
__global__ void k_final(const float* __restrict__ pool, const float* __restrict__ cnt,
                        const float* __restrict__ Wf, const float* __restrict__ bf,
                        float* __restrict__ out) {
    const int g = blockIdx.x;
    const int lane = threadIdx.x;                  // 32 threads = one wave
    float s = 0.0f;
#pragma unroll
    for (int j = 0; j < 4; ++j) {
        int f = lane * 4 + j;
        s += pool[g * FD + f] * Wf[f];
    }
#pragma unroll
    for (int off = 16; off > 0; off >>= 1) s += __shfl_xor(s, off, 32);
    if (lane == 0) out[g] = s / fmaxf(cnt[g], 1.0f) + bf[0];
}

extern "C" void kernel_launch(void* const* d_in, const int* in_sizes, int n_in,
                              void* d_out, int out_size, void* d_ws, size_t ws_size,
                              hipStream_t stream) {
    const float* x     = (const float*)d_in[0];
    const int*   src   = (const int*)d_in[1];
    const int*   dst   = src + NE;
    const int*   batch = (const int*)d_in[2];
    const float* W1    = (const float*)d_in[3];
    const float* b1    = (const float*)d_in[4];
    const float* W2    = (const float*)d_in[5];
    const float* b2    = (const float*)d_in[6];
    const float* Wf    = (const float*)d_in[7];
    const float* bf    = (const float*)d_in[8];
    float* out = (float*)d_out;

    // workspace carve (all re-initialized every call; no cross-call state)
    char* w = (char*)d_ws;
    auto carve = [&](size_t bytes) -> void* {
        void* p = (void*)w;
        w += (bytes + 255) & ~(size_t)255;
        return p;
    };
    float*    dinv = (float*)carve((size_t)NN * 4);
    float*    Gbuf = (float*)carve((size_t)NN * FD * 4);
    float*    agg  = (float*)carve((size_t)NN * FD * 4);
    _Float16* Hb   = (_Float16*)carve((size_t)NN * FD * 2);
    _Float16* W1t  = (_Float16*)carve((size_t)FD * FD * 2);
    _Float16* W2t  = (_Float16*)carve((size_t)FD * FD * 2);
    float*    pool = (float*)carve((size_t)NG * FD * 4);
    float*    cnt  = (float*)carve((size_t)NG * 4);

    const int NF  = NN * FD;                 // 12.8M
    const int B   = 256;
    const int gN  = (NN + B - 1) / B;
    const int gE  = (NE + B - 1) / B;
    const int gNF = (NF + B - 1) / B;

    // degrees -> dinv  (self-loop = init deg to 1)
    k_fill1<<<gN, B, 0, stream>>>(dinv, NN);
    k_deg<<<gE, B, 0, stream>>>(dst, dinv);
    k_rsqrt<<<gN, B, 0, stream>>>(dinv, NN);

    // f16 operands
    k_cast_f16<<<gNF, B, 0, stream>>>(x, Hb, NF);
    k_cast_wT<<<(FD * FD) / B, B, 0, stream>>>(W1, W1t);
    k_cast_wT<<<(FD * FD) / B, B, 0, stream>>>(W2, W2t);

    // pooling accumulators
    k_zero<<<(NG * FD) / B, B, 0, stream>>>(pool, NG * FD);
    k_zero<<<1, B, 0, stream>>>(cnt, NG);
    k_count<<<gN, B, 0, stream>>>(batch, cnt);

    const int gemmBlocks = (NN / 16) * (FD / 16) / 8;   // 6250, one wave per 16x16 tile
    const int scatBlocks = NE / 8;                      // one wave per edge

    // ---- layer 1 ----
    k_gemm_scaled<<<gemmBlocks, B, 0, stream>>>(Hb, W1t, dinv, Gbuf);
    k_copy4<<<(NF / 4) / B, B, 0, stream>>>((const float4*)Gbuf, (float4*)agg, NF / 4); // self-loops
    k_scatter<<<scatBlocks, B, 0, stream>>>(src, dst, Gbuf, agg);
    k_epi_relu_cast<<<gNF, B, 0, stream>>>(agg, dinv, b1, Hb);

    // ---- layer 2 ----
    k_gemm_scaled<<<gemmBlocks, B, 0, stream>>>(Hb, W2t, dinv, Gbuf);
    k_copy4<<<(NF / 4) / B, B, 0, stream>>>((const float4*)Gbuf, (float4*)agg, NF / 4);
    k_scatter<<<scatBlocks, B, 0, stream>>>(src, dst, Gbuf, agg);
    k_epi_pool<<<gNF, B, 0, stream>>>(agg, dinv, b2, batch, pool);

    // ---- readout ----
    k_final<<<NG, 32, 0, stream>>>(pool, cnt, Wf, bf, out);
}
